// AttnPlainNet_81192061764210
// MI455X (gfx1250) — matrix operand order repaired and checked by reading, not compiled, not run on hardware
//
#include <hip/hip_runtime.h>
#include <hip/hip_bf16.h>

// ---------------------------------------------------------------------------
// AttnPlainNet fused implementation for gfx1250 (MI455X, wave32, WMMA).
//
// Shapes: B=256, N=32, C=1, F=512, H1=H2=16, NCLS=64.
// C==1 collapses l2norm to sign(); fadj[b,a,e] = s_a*s_e*t(u_a+u_e)/D[e]
// with t(v)=sign(v)*sqrt(|v|).  fadj (256MB) is regenerated on the fly from
// per-batch vectors s,u,1/D instead of being stored (saves ~22us of HBM).
// Big GEMMs run on v_wmma_f32_16x16x32_f16; fragments use 128-bit LDS/VMEM.
// ---------------------------------------------------------------------------

typedef __attribute__((ext_vector_type(16))) _Float16 v16h;
typedef __attribute__((ext_vector_type(8)))  _Float16 v8h;
typedef __attribute__((ext_vector_type(8)))  float    v8f;
typedef __attribute__((ext_vector_type(4)))  float    v4f;

constexpr int kB = 256, kN = 32, kF = 512, kH = 16, kCLS = 64;
constexpr int kKF = kH * kF;            // 8192 flattened features for classifier

// workspace layout (float elements)
constexpr size_t OFF_S   = 0;                         // [B,F] signs of x
constexpr size_t OFF_U   = OFF_S   + (size_t)kB * kF; // [B,F] u = s*ybar
constexpr size_t OFF_RD  = OFF_U   + (size_t)kB * kF; // [B,F] 1/(D+eps_row)
constexpr size_t OFF_Z1  = OFF_RD  + (size_t)kB * kF; // [B,F] z1
constexpr size_t OFF_H2  = OFF_Z1  + (size_t)kB * kF; // [B,H2,F] h2 pre-BN
constexpr size_t OFF_Z1P = OFF_H2  + (size_t)kB * kH * kF; // [B][2] z1 partials
constexpr size_t OFF_H2P = OFF_Z1P + (size_t)kB * 2;       // [B][32] h2 partials
// total = 2,630,144 floats = 10.52 MB  (must fit in ws_size)

__device__ __forceinline__ float block_reduce_256(float v, float* red, int tid) {
  red[tid] = v; __syncthreads();
  #pragma unroll
  for (int s = 128; s > 0; s >>= 1) {
    if (tid < s) red[tid] += red[tid + s];
    __syncthreads();
  }
  float r = red[0]; __syncthreads();
  return r;
}

__device__ __forceinline__ float softsign_f(float h) { return h / (1.0f + fabsf(h)); }
__device__ __forceinline__ float signf(float v) { return (v >= 0.0f) ? 1.0f : -1.0f; }

// ---------------------------------------------------------------------------
// Kernel 1: per-batch attention + fadj factors (s, u, 1/D) + z1 (+ z1 stats)
// ---------------------------------------------------------------------------
__global__ __launch_bounds__(256) void k1_fadj_stats(
    const float* __restrict__ x, const float* __restrict__ nb,
    const float* __restrict__ w1, const float* __restrict__ w2,
    float* __restrict__ ws)
{
  const int b = blockIdx.x, tid = threadIdx.x;
  __shared__ float sS[kF], sU[kF], sW[kN], red[256], part[256];

  // signs of x and s_x = sum_f s[f]*w1[f]
  float sxp = 0.0f;
  for (int f = tid; f < kF; f += 256) {
    float sg = signf(x[(size_t)b * kF + f]);
    sS[f] = sg; sxp += sg * w1[f];
  }
  const float s_x = block_reduce_256(sxp, red, tid);

  // s_y[n] = sum_f sign(nb[n,f])*w2[f]  (8 deterministic partials per n)
  { int n = tid >> 3, seg = tid & 7;
    const float* nr = nb + ((size_t)(b * kN + n)) * kF + seg * 64;
    const float* wr = w2 + seg * 64;
    float p = 0.0f;
    #pragma unroll 8
    for (int j = 0; j < 64; ++j) p += signf(nr[j]) * wr[j];
    part[tid] = p; }
  __syncthreads();
  if (tid < kN) { float s = 0.0f;
    for (int j = 0; j < 8; ++j) s += part[tid * 8 + j];
    sW[tid] = s_x * s; }                               // attention logits
  __syncthreads();
  if (tid == 0) {                                      // softmax over 32 neighbors
    float mx = sW[0];
    for (int n = 1; n < kN; ++n) mx = fmaxf(mx, sW[n]);
    float ss = 0.0f;
    for (int n = 0; n < kN; ++n) { float e = expf(sW[n] - mx); sW[n] = e; ss += e; }
    float inv = 1.0f / ss;
    for (int n = 0; n < kN; ++n) sW[n] *= inv;
  }
  __syncthreads();

  // u[e] = s[e] * sum_n w[n]*sign(nb[n,e])
  for (int e = tid; e < kF; e += 256) {
    float acc = 0.0f;
    #pragma unroll
    for (int n = 0; n < kN; ++n)
      acc += sW[n] * signf(nb[((size_t)(b * kN + n)) * kF + e]);
    float uu = sS[e] * acc;
    sU[e] = uu;
    ws[OFF_U + (size_t)b * kF + e] = uu;
    ws[OFF_S + (size_t)b * kF + e] = sS[e];
  }
  __syncthreads();

  // D[e] = sum_a sqrt|u_a+u_e|;  z1[e] = s_e * sum_a t(u_a+u_e) / (D+eps)
  float ps = 0.0f, pss = 0.0f;
  for (int e = tid; e < kF; e += 256) {
    const float ue = sU[e];
    float D = 0.0f, Z = 0.0f;
    #pragma unroll 4
    for (int a = 0; a < kF; ++a) {
      float v  = sU[a] + ue;
      float sq = sqrtf(fabsf(v));        // v_sqrt_f32 (TRANS)
      D += sq;
      Z  = (v >= 0.0f) ? (Z + sq) : (Z - sq);
    }
    float rd = 1.0f / (D + 1e-7f);
    float z1 = sS[e] * Z * rd;
    ws[OFF_RD + (size_t)b * kF + e] = rd;
    ws[OFF_Z1 + (size_t)b * kF + e] = z1;
    ps += z1; pss += z1 * z1;
  }
  float tS = block_reduce_256(ps,  red, tid);
  float tQ = block_reduce_256(pss, red, tid);
  if (tid == 0) { ws[OFF_Z1P + b * 2] = tS; ws[OFF_Z1P + b * 2 + 1] = tQ; }
}

// ---------------------------------------------------------------------------
// Kernel 2: BN1+softsign -> h1 (LDS f16); z2 = h1 @ fadj via WMMA with
// B-fragments synthesized from (s,u,1/D); h2pre = W2@z2 + b2; channel stats.
// ---------------------------------------------------------------------------
// pad +8 halves: row stride 1040B = 260 dwords = 4 banks/row -> the 16 A rows
// occupy disjoint bank quads for the 16-byte fragment loads.
constexpr int kHPad = kF + 8;

__global__ __launch_bounds__(256) void k2_gcn_layers(
    const float* __restrict__ W1, const float* __restrict__ b1,
    const float* __restrict__ g1, const float* __restrict__ be1,
    const float* __restrict__ W2, const float* __restrict__ b2,
    float* __restrict__ ws)
{
  const int b = blockIdx.x, tid = threadIdx.x;
  __shared__ alignas(16) float sU[kF];
  __shared__ alignas(16) float sSg[kF];
  __shared__ float sRD[kF], sZ1[kF];
  __shared__ float red[256], sW2[kH * kH];
  __shared__ alignas(16) _Float16 hA[kH * kHPad];  // h1, A-matrix source
  __shared__ alignas(16) float sZ2[kH * kF];       // z2 (f32)

  // global z1 stats from deterministic per-block partials
  float aS = ws[OFF_Z1P + tid * 2], aQ = ws[OFF_Z1P + tid * 2 + 1];
  const float S = block_reduce_256(aS, red, tid);
  const float Q = block_reduce_256(aQ, red, tid);
  const float invBF = 1.0f / ((float)kB * (float)kF);
  const float m = S * invBF;
  const float v = Q * invBF - m * m;

  for (int f = tid; f < kF; f += 256) {
    size_t o = (size_t)b * kF + f;
    sU[f]  = ws[OFF_U  + o];
    sSg[f] = ws[OFF_S  + o];
    sRD[f] = ws[OFF_RD + o];
    sZ1[f] = ws[OFF_Z1 + o];
  }
  sW2[tid] = W2[tid];   // 16x16
  __syncthreads();

  // h1[o,e] = softsign( W1[o]*(z1-m)*rsqrt(W1^2 v + eps)*g1[o] + be1[o] )
  for (int i = tid; i < kH * kF; i += 256) {
    int o = i >> 9, e = i & (kF - 1);
    float W1o = W1[o];
    float inv = rsqrtf(W1o * W1o * v + 1e-5f);
    float h = (sZ1[e] - m) * W1o * inv * g1[o] + be1[o];
    hA[o * kHPad + e] = (_Float16)softsign_f(h);
  }
  __syncthreads();

  const int lane = tid & 31, wave = tid >> 5, hi = lane >> 4, Mn = lane & 15;
  for (int t8 = 0; t8 < 4; ++t8) {
    const int n0   = (wave + t8 * 8) * 16;     // e-tile
    const int ecol = n0 + Mn;                  // this lane's column (N = lane%16)
    const float ue = sU[ecol];
    const float srd = sSg[ecol] * sRD[ecol];   // s_e / (D_e + eps)
    v8f acc = {};
    for (int kb = 0; kb < kF / 32; ++kb) {
      const int k0 = kb * 32;
      // ---- A fragment: two contiguous 16B runs per lane (ds_load_b128) ----
      const v8h* arow = (const v8h*)(hA + Mn * kHPad + k0);
      v8h a0 = arow[2 * hi];       // K = k0 + 16*hi + 0..7
      v8h a1 = arow[2 * hi + 1];   // K = k0 + 16*hi + 8..15
      v16h af = __builtin_shufflevector(a0, a1, 0, 1, 2, 3, 4, 5, 6, 7,
                                        8, 9, 10, 11, 12, 13, 14, 15);
      // ---- B fragment: synthesize fadj column from u,s (128-bit LDS reads) --
      const v4f* up = (const v4f*)(sU  + k0 + (hi << 4));
      const v4f* sp = (const v4f*)(sSg + k0 + (hi << 4));
      v16h bf;
      #pragma unroll
      for (int q = 0; q < 4; ++q) {
        v4f uq = up[q], sq4 = sp[q];
        #pragma unroll
        for (int t = 0; t < 4; ++t) {
          float vv = uq[t] + ue;
          float sq = sqrtf(fabsf(vv));
          float val = ((vv >= 0.0f) ? sq : -sq) * sq4[t] * srd;
          bf[q * 4 + t] = (_Float16)val;
        }
      }
      acc = __builtin_amdgcn_wmma_f32_16x16x32_f16(
          false, af, false, bf, (short)0, acc, false, false);
    }
    #pragma unroll
    for (int r = 0; r < 8; ++r)                // C layout: M = r + 8*hi, N = Mn
      sZ2[(r + (hi << 3)) * kF + ecol] = acc[r];
  }
  __syncthreads();

  // h2pre[o,e] = b2[o] + sum_c W2[o,c]*z2[c,e];  per-channel stats partials
  { const int o = tid >> 4, l16 = tid & 15;
    const float b2o = b2[o];
    float ps = 0.0f, pss = 0.0f;
    for (int j = 0; j < kF / 16; ++j) {
      int e = l16 + j * 16;
      float a2 = b2o;
      #pragma unroll
      for (int c = 0; c < 16; ++c) a2 += sW2[o * 16 + c] * sZ2[c * kF + e];
      ws[OFF_H2 + ((size_t)(b * kH + o)) * kF + e] = a2;
      ps += a2; pss += a2 * a2;
    }
    red[tid] = ps; __syncthreads();
    if (l16 == 0) { float s = 0.0f;
      for (int j = 0; j < 16; ++j) s += red[o * 16 + j];
      ws[OFF_H2P + (size_t)blockIdx.x * 32 + o] = s; }
    __syncthreads();
    red[tid] = pss; __syncthreads();
    if (l16 == 0) { float s = 0.0f;
      for (int j = 0; j < 16; ++j) s += red[o * 16 + j];
      ws[OFF_H2P + (size_t)blockIdx.x * 32 + 16 + o] = s; }
  }
}

// ---------------------------------------------------------------------------
// Kernel 3: finalize BN2 stats; out = softsign(BN2(h2)).reshape(B,8192) @ Wc^T + bc
// 16 blocks x 128 threads (4 waves); each wave owns a 16-class N-tile.
// ---------------------------------------------------------------------------
__global__ __launch_bounds__(128) void k3_classifier(
    const float* __restrict__ g2, const float* __restrict__ be2,
    const float* __restrict__ Wc, const float* __restrict__ bc,
    const float* __restrict__ ws, float* __restrict__ out)
{
  const int tid = threadIdx.x, m0 = blockIdx.x * 16;
  __shared__ float sM[kH], sA[kH], sB[kH], red[128], red2[128];

  // reduce 256 per-block channel partials -> mean/var per channel
  { int o = tid >> 3, s8 = tid & 7;
    float ps = 0.0f, pss = 0.0f;
    for (int j = 0; j < 32; ++j) {
      int blk = s8 + j * 8;
      ps  += ws[OFF_H2P + (size_t)blk * 32 + o];
      pss += ws[OFF_H2P + (size_t)blk * 32 + 16 + o];
    }
    red[tid] = ps; red2[tid] = pss; }
  __syncthreads();
  if ((tid & 7) == 0) {
    int o = tid >> 3;
    float s = 0.0f, q = 0.0f;
    for (int j = 0; j < 8; ++j) { s += red[tid + j]; q += red2[tid + j]; }
    const float invBF = 1.0f / ((float)kB * (float)kF);
    float m = s * invBF, v = q * invBF - m * m;
    sM[o] = m; sA[o] = g2[o] * rsqrtf(v + 1e-5f); sB[o] = be2[o];
  }
  __syncthreads();

  const int lane = tid & 31, wave = tid >> 5, hi = lane >> 4, Mn = lane & 15;
  const int n0 = wave * 16;
  const float* hrow = ws + OFF_H2 + (size_t)(m0 + Mn) * kKF;  // A row = batch
  const float* wrow = Wc + (size_t)(n0 + Mn) * kKF;           // B col = class
  v8f acc = {};
  for (int kb = 0; kb < kKF / 32; ++kb) {
    const int k0 = kb * 32;
    // o = K>>9 is uniform over the whole 32-wide K-block (k0 % 32 == 0)
    const int o = k0 >> 9;
    const float mo = sM[o], ao = sA[o], bo = sB[o];
    // ---- A fragment: BN2+softsign fused on 4x128-bit global loads ----
    const v4f* ap0 = (const v4f*)(hrow + k0 + (hi << 4));        // K=k0+16hi+0..7
    const v4f* ap1 = (const v4f*)(hrow + k0 + 8 + (hi << 4));    // K=k0+16hi+8..15
    v4f h0 = ap0[0], h1 = ap0[1], h2 = ap1[0], h3 = ap1[1];
    v16h af;
    #pragma unroll
    for (int t = 0; t < 4; ++t) {
      af[t]      = (_Float16)softsign_f((h0[t] - mo) * ao + bo);
      af[t + 4]  = (_Float16)softsign_f((h1[t] - mo) * ao + bo);
      af[t + 8]  = (_Float16)softsign_f((h2[t] - mo) * ao + bo);
      af[t + 12] = (_Float16)softsign_f((h3[t] - mo) * ao + bo);
    }
    // ---- B fragment: 16 consecutive Wc floats per lane (K = k0+16hi+t) ----
    const v4f* wp = (const v4f*)(wrow + k0 + (hi << 4));
    v16h bf;
    #pragma unroll
    for (int q = 0; q < 4; ++q) {
      v4f wq = wp[q];
      #pragma unroll
      for (int t = 0; t < 4; ++t) bf[q * 4 + t] = (_Float16)wq[t];
    }
    acc = __builtin_amdgcn_wmma_f32_16x16x32_f16(
        false, af, false, bf, (short)0, acc, false, false);
  }
  const float bcv = bc[n0 + Mn];
  #pragma unroll
  for (int r = 0; r < 8; ++r)
    out[(size_t)(m0 + r + (hi << 3)) * kCLS + n0 + Mn] = acc[r] + bcv;
}

// ---------------------------------------------------------------------------
extern "C" void kernel_launch(void* const* d_in, const int* in_sizes, int n_in,
                              void* d_out, int out_size, void* d_ws, size_t ws_size,
                              hipStream_t stream) {
  const float* x    = (const float*)d_in[0];
  const float* nb   = (const float*)d_in[1];
  const float* a1w  = (const float*)d_in[2];
  const float* a2w  = (const float*)d_in[3];
  const float* W1   = (const float*)d_in[4];
  const float* b1   = (const float*)d_in[5];
  const float* g1   = (const float*)d_in[6];
  const float* be1  = (const float*)d_in[7];
  const float* W2   = (const float*)d_in[8];
  const float* b2   = (const float*)d_in[9];
  const float* g2   = (const float*)d_in[10];
  const float* be2  = (const float*)d_in[11];
  const float* Wc   = (const float*)d_in[12];
  const float* bc   = (const float*)d_in[13];
  float* out = (float*)d_out;
  float* ws  = (float*)d_ws;

  k1_fadj_stats<<<kB, 256, 0, stream>>>(x, nb, a1w, a2w, ws);
  k2_gcn_layers<<<kB, 256, 0, stream>>>(W1, b1, g1, be1, W2, b2, ws);
  k3_classifier<<<kB / 16, 128, 0, stream>>>(g2, be2, Wc, bc, ws, out);
}
